// MixtureOfDepths_27685359190825
// MI455X (gfx1250) — compile-verified, matrix operand order checked by reading, not run on hardware
//
#include <hip/hip_runtime.h>

// ---------------- problem constants (match reference) ----------------
static constexpr int kB   = 4;
static constexpr int kS   = 4096;
static constexpr int kD   = 2048;
static constexpr int kDQ  = 512;     // router hidden
static constexpr int kDF  = 8192;    // FFN hidden
static constexpr int kK   = 512;     // capacity 0.125 * S
static constexpr int kBK  = kB * kK; // 2048 selected rows total
static constexpr int kBS  = kB * kS; // 16384 tokens

typedef __attribute__((ext_vector_type(2)))  float  v2f;
typedef __attribute__((ext_vector_type(8)))  float  v8f;
typedef __attribute__((ext_vector_type(16))) __bf16 v16bf;

// =====================================================================
// Transpose: in[R][C] -> out[C][R], optionally converting to bf16.
// R, C multiples of 32. block = (32,8).
// =====================================================================
template <typename TOUT>
__global__ __launch_bounds__(256) void transpose_kernel(
    const float* __restrict__ in, TOUT* __restrict__ out, int R, int C) {
  __shared__ float tile[32][33];
  const int c0 = blockIdx.x * 32;
  const int r0 = blockIdx.y * 32;
  const int tx = threadIdx.x, ty = threadIdx.y;
  for (int i = 0; i < 32; i += 8)
    tile[ty + i][tx] = in[(size_t)(r0 + ty + i) * C + (c0 + tx)];
  __syncthreads();
  for (int i = 0; i < 32; i += 8)
    out[(size_t)(c0 + ty + i) * R + (r0 + tx)] = (TOUT)tile[tx][ty + i];
}

// =====================================================================
// Passthrough copy of x into output (float4 vectorized).
// =====================================================================
__global__ __launch_bounds__(256) void copy_x_kernel(
    const float4* __restrict__ x, float4* __restrict__ out, int n4) {
  int i = blockIdx.x * 256 + threadIdx.x;
  if (i < n4) out[i] = x[i];
}

// =====================================================================
// Router: scores[m] = relu(x[m,:] @ Wr1 + br1) @ Wr2 + br2
// fp32 WMMA 16x16x4. One block = 128 token rows, 256 threads = 8 waves
// arranged 2 (M) x 4 (N); each wave owns a 64x32 accumulator region.
// Wr1t is pre-transposed: [DQ][D].
// =====================================================================
__global__ __launch_bounds__(256) void router_kernel(
    const float* __restrict__ x, const float* __restrict__ Wr1t,
    const float* __restrict__ br1, const float* __restrict__ Wr2,
    const float* __restrict__ br2, float* __restrict__ scores) {
  __shared__ float As[128][36];   // 144 B pitch: 16B aligned rows
  __shared__ float Bs[128][36];
  __shared__ float rowScore[128];

  const int tid   = threadIdx.x;
  const int lane  = tid & 31;
  const int wave  = tid >> 5;
  const int waveM = wave >> 2;     // 0..1
  const int waveN = wave & 3;      // 0..3
  const int half  = lane >> 4;     // 0..1
  const int l16   = lane & 15;
  const int m0    = blockIdx.x * 128;

  if (tid < 128) rowScore[tid] = 0.0f;
  __syncthreads();

  const v8f vzero = {0.f, 0.f, 0.f, 0.f, 0.f, 0.f, 0.f, 0.f};

  for (int n0 = 0; n0 < kDQ; n0 += 128) {
    v8f acc[4][2];
    for (int a = 0; a < 4; ++a)
      for (int b = 0; b < 2; ++b) acc[a][b] = vzero;

    for (int k0 = 0; k0 < kD; k0 += 32) {
      // stage 128x32 fp32 tiles (coalesced float4 loads)
      const int colBase = (tid & 7) * 4;
      const int rowBase = tid >> 3;  // 0..31
      for (int rr = 0; rr < 128; rr += 32) {
        *(float4*)&As[rowBase + rr][colBase] =
            *(const float4*)&x[(size_t)(m0 + rowBase + rr) * kD + k0 + colBase];
        *(float4*)&Bs[rowBase + rr][colBase] =
            *(const float4*)&Wr1t[(size_t)(n0 + rowBase + rr) * kD + k0 + colBase];
      }
      __syncthreads();

      for (int ks = 0; ks < 8; ++ks) {   // 8 x K=4 steps per staged tile
        v2f a2[4], b2[2];
        for (int wm = 0; wm < 4; ++wm) {
          int m = waveM * 64 + wm * 16 + l16;
          a2[wm] = *(const v2f*)&As[m][ks * 4 + 2 * half];
        }
        for (int wn = 0; wn < 2; ++wn) {
          int n = waveN * 32 + wn * 16 + l16;
          b2[wn] = *(const v2f*)&Bs[n][ks * 4 + 2 * half];
        }
        for (int wm = 0; wm < 4; ++wm)
          for (int wn = 0; wn < 2; ++wn)
            acc[wm][wn] = __builtin_amdgcn_wmma_f32_16x16x4_f32(
                false, a2[wm], false, b2[wn], (short)0, acc[wm][wn],
                false, false);
      }
      __syncthreads();
    }

    // epilogue: h = relu(acc + br1); partial score += h * Wr2[n]
    for (int wm = 0; wm < 4; ++wm) {
      for (int wn = 0; wn < 2; ++wn) {
        const int nglob = n0 + waveN * 32 + wn * 16 + l16;
        const float w2 = Wr2[nglob];
        const float b1 = br1[nglob];
        for (int r = 0; r < 8; ++r) {
          float h = acc[wm][wn][r] + b1;
          h = fmaxf(h, 0.0f);
          float p = h * w2;
          p += __shfl_xor(p, 1, 16);
          p += __shfl_xor(p, 2, 16);
          p += __shfl_xor(p, 4, 16);
          p += __shfl_xor(p, 8, 16);
          if (l16 == 0) {
            int row = waveM * 64 + wm * 16 + r + 8 * half;
            atomicAdd(&rowScore[row], p);
          }
        }
      }
    }
    __syncthreads();
  }

  if (tid < 128) scores[m0 + tid] = rowScore[tid] + br2[0];
}

// =====================================================================
// Top-k per batch row: bitonic sort of 4096 (score,idx) keys in LDS.
// block = 512 threads, grid = B. Writes index list + float mask row.
// =====================================================================
__global__ __launch_bounds__(512) void topk_kernel(
    const float* __restrict__ scores, int* __restrict__ topkIdx,
    float* __restrict__ maskOut) {
  __shared__ unsigned long long keys[kS];
  const int b = blockIdx.x;
  const int tid = threadIdx.x;

  for (int i = tid; i < kS; i += 512) {
    unsigned u = __float_as_uint(scores[b * kS + i]);
    u = (u & 0x80000000u) ? ~u : (u | 0x80000000u);  // monotonic key
    keys[i] = ((unsigned long long)u << 32) | (unsigned)i;
  }
  __syncthreads();

  for (int size = 2; size <= kS; size <<= 1) {
    for (int stride = size >> 1; stride > 0; stride >>= 1) {
      for (int t = tid; t < kS / 2; t += 512) {
        int i = 2 * t - (t & (stride - 1));
        int j = i + stride;
        bool dirDesc = ((i & size) == 0);
        unsigned long long a = keys[i], c = keys[j];
        if ((a < c) == dirDesc) { keys[i] = c; keys[j] = a; }
      }
      __syncthreads();
    }
  }

  for (int i = tid; i < kS; i += 512) maskOut[b * kS + i] = 0.0f;
  __syncthreads();
  if (tid < kK) {
    int idx = (int)(keys[tid] & 0xFFFFFFFFu);
    topkIdx[b * kK + tid] = idx;
    maskOut[b * kS + idx] = 1.0f;
  }
}

// =====================================================================
// Gather selected token rows -> bf16 matrix selA[BK][D].
// =====================================================================
__global__ __launch_bounds__(256) void gather_kernel(
    const float* __restrict__ x, const int* __restrict__ topkIdx,
    __bf16* __restrict__ selA) {
  const int r = blockIdx.x;
  const int b = r >> 9;  // r / 512
  const int tok = topkIdx[r];
  const float* src = x + ((size_t)b * kS + tok) * kD;
  __bf16* dst = selA + (size_t)r * kD;
  const int c = threadIdx.x * 8;  // 256 * 8 == 2048 == D
  float4 f0 = *(const float4*)&src[c];
  float4 f1 = *(const float4*)&src[c + 4];
  union { __bf16 h[8]; uint4 q; } u;
  u.h[0] = (__bf16)f0.x; u.h[1] = (__bf16)f0.y;
  u.h[2] = (__bf16)f0.z; u.h[3] = (__bf16)f0.w;
  u.h[4] = (__bf16)f1.x; u.h[5] = (__bf16)f1.y;
  u.h[6] = (__bf16)f1.z; u.h[7] = (__bf16)f1.w;
  *(uint4*)&dst[c] = u.q;
}

// =====================================================================
// bf16 GEMM: C[M][N] = A[M][K] * Bt[N][K]^T  (+bias, epilogue variant)
// v_wmma_f32_16x16x32_bf16; block tile 128x128x32; 8 waves (2x4),
// each wave: 4x2 grid of 16x16 f32 accumulators (64x32 region).
// EPI=0: relu -> bf16 store to outH[M][N]
// EPI=1: fp32 scatter to outX at row (b = m/512, token = topkIdx[m])
// =====================================================================
template <int EPI>
__global__ __launch_bounds__(256) void gemm_bf16_kernel(
    const __bf16* __restrict__ A, const __bf16* __restrict__ Bt,
    const float* __restrict__ bias, int M, int N, int K,
    __bf16* __restrict__ outH, float* __restrict__ outX,
    const int* __restrict__ topkIdx) {
  __shared__ __bf16 As[128][40];  // 80 B pitch (16B aligned)
  __shared__ __bf16 Bs[128][40];

  const int tid   = threadIdx.x;
  const int lane  = tid & 31;
  const int wave  = tid >> 5;
  const int waveM = wave >> 2;  // 0..1
  const int waveN = wave & 3;   // 0..3
  const int half  = lane >> 4;
  const int l16   = lane & 15;
  const int n0    = blockIdx.x * 128;
  const int m0    = blockIdx.y * 128;

  const v8f vzero = {0.f, 0.f, 0.f, 0.f, 0.f, 0.f, 0.f, 0.f};
  v8f acc[4][2];
  for (int a = 0; a < 4; ++a)
    for (int b = 0; b < 2; ++b) acc[a][b] = vzero;

  for (int k0 = 0; k0 < K; k0 += 32) {
    // stage: 128 rows x 32 bf16 = 512 chunks of 16B; 2 chunks/thread
    for (int c = tid; c < 512; c += 256) {
      int r = c >> 2, col = (c & 3) * 8;
      *(uint4*)&As[r][col] = *(const uint4*)&A[(size_t)(m0 + r) * K + k0 + col];
      *(uint4*)&Bs[r][col] = *(const uint4*)&Bt[(size_t)(n0 + r) * K + k0 + col];
    }
    if (k0 + 32 < K) {  // prefetch next K tile -> global_prefetch_b8
      __builtin_prefetch(&A[(size_t)(m0 + (tid >> 2)) * K + k0 + 32], 0, 0);
      __builtin_prefetch(&Bt[(size_t)(n0 + (tid >> 2)) * K + k0 + 32], 0, 0);
    }
    __syncthreads();

    v16bf av[4], bv[2];
    union U { uint4 q[2]; v16bf v; } u;
    for (int wm = 0; wm < 4; ++wm) {
      int m = waveM * 64 + wm * 16 + l16;
      u.q[0] = *(const uint4*)&As[m][8 * half];
      u.q[1] = *(const uint4*)&As[m][16 + 8 * half];
      av[wm] = u.v;
    }
    for (int wn = 0; wn < 2; ++wn) {
      int n = waveN * 32 + wn * 16 + l16;
      u.q[0] = *(const uint4*)&Bs[n][8 * half];
      u.q[1] = *(const uint4*)&Bs[n][16 + 8 * half];
      bv[wn] = u.v;
    }
    for (int wm = 0; wm < 4; ++wm)
      for (int wn = 0; wn < 2; ++wn)
        acc[wm][wn] = __builtin_amdgcn_wmma_f32_16x16x32_bf16(
            false, av[wm], false, bv[wn], (short)0, acc[wm][wn],
            false, false);
    __syncthreads();
  }

  for (int wm = 0; wm < 4; ++wm) {
    const int mloc = waveM * 64 + wm * 16;
    for (int wn = 0; wn < 2; ++wn) {
      const int n = n0 + waveN * 32 + wn * 16 + l16;
      const float bb = bias[n];
      for (int r = 0; r < 8; ++r) {
        const int m = m0 + mloc + r + 8 * half;
        float v = acc[wm][wn][r] + bb;
        if (EPI == 0) {
          v = fmaxf(v, 0.0f);
          outH[(size_t)m * N + n] = (__bf16)v;
        } else {
          const int bidx = m >> 9;
          const int tok = topkIdx[m];
          outX[((size_t)bidx * kS + tok) * (size_t)kD + n] = v;
        }
      }
    }
  }
}

// =====================================================================
// launch
// =====================================================================
extern "C" void kernel_launch(void* const* d_in, const int* in_sizes, int n_in,
                              void* d_out, int out_size, void* d_ws,
                              size_t ws_size, hipStream_t stream) {
  (void)in_sizes; (void)n_in; (void)out_size; (void)ws_size;
  const float* x   = (const float*)d_in[0];
  const float* Wr1 = (const float*)d_in[1];
  const float* br1 = (const float*)d_in[2];
  const float* Wr2 = (const float*)d_in[3];
  const float* br2 = (const float*)d_in[4];
  const float* Wf1 = (const float*)d_in[5];
  const float* bf1 = (const float*)d_in[6];
  const float* Wf2 = (const float*)d_in[7];
  const float* bf2 = (const float*)d_in[8];

  float* out  = (float*)d_out;
  float* mask = out + (size_t)kBS * kD;

  // workspace layout
  char* wsb = (char*)d_ws;
  size_t off = 0;
  auto alloc = [&](size_t bytes) {
    void* p = wsb + off;
    off = (off + bytes + 255) & ~(size_t)255;
    return p;
  };
  float*  scores = (float*) alloc((size_t)kBS * 4);
  int*    topk   = (int*)   alloc((size_t)kBK * 4);
  float*  Wr1t   = (float*) alloc((size_t)kDQ * kD * 4);
  __bf16* Wf1t   = (__bf16*)alloc((size_t)kDF * kD * 2);
  __bf16* Wf2t   = (__bf16*)alloc((size_t)kD * kDF * 2);
  __bf16* selA   = (__bf16*)alloc((size_t)kBK * kD * 2);
  __bf16* H      = (__bf16*)alloc((size_t)kBK * kDF * 2);

  const dim3 tb(32, 8);
  // weight pre-transposes (+ bf16 conversion for FFN weights)
  transpose_kernel<float><<<dim3(kDQ / 32, kD / 32), tb, 0, stream>>>(
      Wr1, Wr1t, kD, kDQ);
  transpose_kernel<__bf16><<<dim3(kDF / 32, kD / 32), tb, 0, stream>>>(
      Wf1, Wf1t, kD, kDF);
  transpose_kernel<__bf16><<<dim3(kD / 32, kDF / 32), tb, 0, stream>>>(
      Wf2, Wf2t, kDF, kD);

  // passthrough copy x -> out
  const int n4 = kBS * kD / 4;
  copy_x_kernel<<<(n4 + 255) / 256, 256, 0, stream>>>(
      (const float4*)x, (float4*)out, n4);

  // router (fp32 WMMA) -> scores
  router_kernel<<<kBS / 128, 256, 0, stream>>>(x, Wr1t, br1, Wr2, br2, scores);

  // top-k + mask
  topk_kernel<<<kB, 512, 0, stream>>>(scores, topk, mask);

  // gather selected rows -> bf16
  gather_kernel<<<kBK, 256, 0, stream>>>(x, topk, selA);

  // FFN GEMM1: H = relu(selA @ Wf1 + bf1), bf16
  gemm_bf16_kernel<0><<<dim3(kDF / 128, kBK / 128), 256, 0, stream>>>(
      selA, Wf1t, bf1, kBK, kDF, kD, H, nullptr, nullptr);

  // FFN GEMM2: out[sel rows] = H @ Wf2 + bf2, fp32 scatter
  gemm_bf16_kernel<1><<<dim3(kD / 128, kBK / 128), 256, 0, stream>>>(
      H, Wf2t, bf2, kBK, kD, kDF, nullptr, out, topk);
}
